// Sampler_59150289600746
// MI455X (gfx1250) — compile-verified, matrix operand order checked by reading, not run on hardware
//
#include <hip/hip_runtime.h>
#include <hip/hip_bf16.h>

#define B_ 256
#define V_ 128000
#define D_ 1024
#define H_ 200
#define CAND 8192

typedef __attribute__((ext_vector_type(8)))  float  v8f;
typedef __attribute__((ext_vector_type(16))) __bf16 v16bf;

// Convert a 16-float LDS run-pair (k0..k0+7, k0+16..k0+23) into a v16bf
// A/B fragment (16-bit WMMA operand layout). rowp must be 16B-aligned.
__device__ __forceinline__ v16bf frag16(const float* rowp, int k0) {
  float4 x0 = *(const float4*)(rowp + k0);
  float4 x1 = *(const float4*)(rowp + k0 + 4);
  float4 x2 = *(const float4*)(rowp + k0 + 16);
  float4 x3 = *(const float4*)(rowp + k0 + 20);
  v16bf r;
  r[0]  = (__bf16)x0.x; r[1]  = (__bf16)x0.y; r[2]  = (__bf16)x0.z; r[3]  = (__bf16)x0.w;
  r[4]  = (__bf16)x1.x; r[5]  = (__bf16)x1.y; r[6]  = (__bf16)x1.z; r[7]  = (__bf16)x1.w;
  r[8]  = (__bf16)x2.x; r[9]  = (__bf16)x2.y; r[10] = (__bf16)x2.z; r[11] = (__bf16)x2.w;
  r[12] = (__bf16)x3.x; r[13] = (__bf16)x3.y; r[14] = (__bf16)x3.z; r[15] = (__bf16)x3.w;
  return r;
}

// ---------------------------------------------------------------------------
// Kernel 1: logits = hidden @ emb^T + bias, bf16 WMMA with f32 accumulate.
// Block = 256 threads (8 waves). Tile: 16 rows (M) x 256 cols (N), K chunk 32.
// Staging uses GLOBAL_LOAD_ASYNC_TO_LDS_B128 (ASYNCcnt) of raw fp32 tiles;
// fp32->bf16 conversion happens at fragment build (v_cvt_pk_bf16_f32).
// 8 consecutive lanes cover one 128B row-chunk -> contiguous async bursts.
// ---------------------------------------------------------------------------
__global__ __launch_bounds__(256) void gemm_kernel(
    const float* __restrict__ hid, const float* __restrict__ emb,
    const float* __restrict__ bias, float* __restrict__ out) {
  // fp32 staging tiles; row stride 36 floats = 144 B (16B aligned, depads banks)
  __shared__ float a_raw[16][36];
  __shared__ float b_raw[256][36];

  const int tid   = threadIdx.x;
  const int vbase = blockIdx.x * 256;
  const int mbase = blockIdx.y * 16;
  const int lane  = tid & 31;
  const int wave  = tid >> 5;
  const int half  = lane >> 4;   // 0: lanes 0-15, 1: lanes 16-31
  const int ln15  = lane & 15;

  const int chunk = tid & 7;     // which 16B chunk of a 128B row slab
  const int rbase = tid >> 3;    // base row (0..31)

  v8f c0 = {};
  v8f c1 = {};

  for (int kb = 0; kb < D_; kb += 32) {
    // --- async-stage B tile: 256 rows x 32 f32. 8 lanes per row => 128B bursts
#pragma unroll
    for (int q = 0; q < 8; ++q) {
      int row = rbase + q * 32;
      const float* gp = emb + (size_t)(vbase + row) * D_ + kb + chunk * 4;
      unsigned lds = (unsigned)(unsigned long long)&b_raw[row][chunk * 4];
      asm volatile("global_load_async_to_lds_b128 %0, %1, off"
                   :: "v"(lds), "v"((unsigned long long)gp)
                   : "memory");
    }
    // --- async-stage A tile: 16 rows x 32 f32 (first 128 threads) ---
    if (tid < 128) {
      const float* gp = hid + (size_t)(mbase + rbase) * D_ + kb + chunk * 4;
      unsigned lds = (unsigned)(unsigned long long)&a_raw[rbase][chunk * 4];
      asm volatile("global_load_async_to_lds_b128 %0, %1, off"
                   :: "v"(lds), "v"((unsigned long long)gp)
                   : "memory");
    }
    if (kb + 32 < D_) {  // keep next slab warm in GL2
      __builtin_prefetch(emb + (size_t)(vbase + rbase) * D_ + kb + 32 + chunk * 4, 0, 1);
    }
    asm volatile("s_wait_asynccnt 0x0" ::: "memory");
    __syncthreads();

    // --- build bf16 fragments from fp32 LDS and issue WMMAs ---
    v16bf a  = frag16(&a_raw[ln15][0], half * 8);
    int r0 = wave * 32 + ln15;
    v16bf b0 = frag16(&b_raw[r0][0], half * 8);
    v16bf b1 = frag16(&b_raw[r0 + 16][0], half * 8);

    c0 = __builtin_amdgcn_wmma_f32_16x16x32_bf16(false, a, false, b0, (short)0,
                                                 c0, false, false);
    c1 = __builtin_amdgcn_wmma_f32_16x16x32_bf16(false, a, false, b1, (short)0,
                                                 c1, false, false);
    __syncthreads();
  }

  // C/D layout: VGPR r, lanes 0-15 -> M=r, lanes 16-31 -> M=r+8; N = ln15
  int col0 = vbase + wave * 32 + ln15;
  int col1 = col0 + 16;
  float bi0 = bias[col0], bi1 = bias[col1];
#pragma unroll
  for (int r = 0; r < 8; ++r) {
    int m = mbase + r + half * 8;
    out[(size_t)m * V_ + col0] = c0[r] + bi0;
    out[(size_t)m * V_ + col1] = c1[r] + bi1;
  }
}

// ---------------------------------------------------------------------------
// Kernel 2: presence/frequency/repetition penalties. One block per row.
// ---------------------------------------------------------------------------
__global__ __launch_bounds__(256) void penalty_kernel(
    float* __restrict__ logits, const int* __restrict__ ids,
    const float* __restrict__ pres, const float* __restrict__ freq,
    const float* __restrict__ rep) {
  int b = blockIdx.x;
  __shared__ int s_ids[H_];
  int t = threadIdx.x;
  if (t < H_) s_ids[t] = ids[b * H_ + t];
  __syncthreads();
  if (t < H_) {
    int id = s_ids[t];
    int count = 0;
    bool first = true;
#pragma unroll 8
    for (int j = 0; j < H_; ++j) {
      int o = s_ids[j];
      count += (o == id) ? 1 : 0;
      if (j < t && o == id) first = false;
    }
    if (first) {
      size_t off = (size_t)b * V_ + id;
      float x = logits[off];
      float rp = rep[b];
      x = (x > 0.f) ? (x / rp) : (x * rp);
      x -= (float)count * freq[b] + pres[b];
      logits[off] = x;
    }
  }
}

// ---------------------------------------------------------------------------
// Kernel 3: per-row top-candidate selection. One block (256 thr) per row.
//  pass1: row max M      pass2: 256-bin histogram + full sum(exp(y-M))
//  pass3: collect (value,idx) candidates >= bucket threshold into LDS
//  bitonic sort 8192 packed u64 keys desc; prefix scans give top-k/top-p/min-p
//  thresholds + final log-sum-exp; gumbel argmax over kept prefix.
// ---------------------------------------------------------------------------
__global__ __launch_bounds__(256) void rowselect_kernel(
    const float* __restrict__ logits, const float* __restrict__ temps,
    const float* __restrict__ topps, const int* __restrict__ topks,
    const float* __restrict__ minps, float* __restrict__ tau,
    float* __restrict__ lse, int* __restrict__ tokens) {
  extern __shared__ unsigned long long cand[];  // CAND u64 keys (64 KB)
  __shared__ float sred[256];
  __shared__ int sidx[256];
  __shared__ int hist[256];
  __shared__ int s_cnt;
  __shared__ float s_f[2];
  __shared__ int s_i[1];

  const int b = blockIdx.x, t = threadIdx.x;
  const float invT = 1.f / temps[b];
  const float* row = logits + (size_t)b * V_;

  // pass 1: max
  float m = -__builtin_inff();
  for (int v = t; v < V_; v += 256) m = fmaxf(m, row[v] * invT);
  sred[t] = m;
  __syncthreads();
  for (int s = 128; s > 0; s >>= 1) {
    if (t < s) sred[t] = fmaxf(sred[t], sred[t + s]);
    __syncthreads();
  }
  const float M = sred[0];
  __syncthreads();

  // pass 2: histogram of (M - y), bin width 1/32, + full softmax denominator
  hist[t] = 0;
  __syncthreads();
  float se = 0.f;
  for (int v = t; v < V_; v += 256) {
    float y = row[v] * invT;
    int bin = (int)((M - y) * 32.f);
    bin = bin < 0 ? 0 : (bin > 255 ? 255 : bin);
    atomicAdd(&hist[bin], 1);
    se += __expf(y - M);
  }
  sred[t] = se;
  __syncthreads();
  for (int s = 128; s > 0; s >>= 1) {
    if (t < s) sred[t] += sred[t + s];
    __syncthreads();
  }
  const float S = sred[0];
  if (t == 0) {
    int cum = 0, tb = 255;
    for (int i = 0; i < 256; ++i) {
      cum += hist[i];
      if (cum >= 1024) { tb = i; break; }
    }
    s_f[0] = M - (float)(tb + 1) * (1.f / 32.f);
    s_cnt = 0;
  }
  __syncthreads();
  const float tcand = s_f[0];

  // pass 3: collect candidates as (orderable-u32(value) << 32 | idx)
  for (int v = t; v < V_; v += 256) {
    float y = row[v] * invT;
    if (y >= tcand) {
      int p = atomicAdd(&s_cnt, 1);
      if (p < CAND) {
        unsigned u = __float_as_uint(y);
        unsigned key = (u & 0x80000000u) ? ~u : (u | 0x80000000u);
        cand[p] = ((unsigned long long)key << 32) | (unsigned)v;
      }
    }
  }
  __syncthreads();
  int n = s_cnt < CAND ? s_cnt : CAND;
  for (int i = n + t; i < CAND; i += 256) cand[i] = 0ull;
  __syncthreads();

  // bitonic sort, descending
  for (int k = 2; k <= CAND; k <<= 1) {
    for (int j = k >> 1; j > 0; j >>= 1) {
      for (int i = t; i < CAND; i += 256) {
        int ixj = i ^ j;
        if (ixj > i) {
          unsigned long long a = cand[i], c = cand[ixj];
          bool desc = ((i & k) == 0);
          if (desc ? (a < c) : (a > c)) { cand[i] = c; cand[ixj] = a; }
        }
      }
      __syncthreads();
    }
  }

  // serial prefix selection (<= top_k <= 1000 iterations)
  if (t == 0) {
    const int K = topks[b];
    const float P = topps[b];
    const float invS = 1.f / S;
    float cum = 0.f, sum1 = 0.f;
    int n1 = 0;
    int lim = (n < K) ? n : K;
    for (int i = 0; i < lim; ++i) {
      if (cum > P) break;  // exclusive-cumsum mask: (cum - p_i) > top_p
      unsigned key = (unsigned)(cand[i] >> 32);
      unsigned u = (key & 0x80000000u) ? (key & 0x7fffffffu) : ~key;
      float y = __uint_as_float(u);
      float e = __expf(y - M);
      cum += e * invS;
      sum1 += e;
      n1++;
    }
    // min-p: keep prefix with y >= M + log(min_p)
    float thr = M + __logf(minps[b]);  // min_p==0 -> -inf -> keeps all
    float sum2 = 0.f, lastY = M;
    int n2 = 0;
    for (int i = 0; i < n1; ++i) {
      unsigned key = (unsigned)(cand[i] >> 32);
      unsigned u = (key & 0x80000000u) ? (key & 0x7fffffffu) : ~key;
      float y = __uint_as_float(u);
      if (y < thr) break;
      sum2 += __expf(y - M);
      lastY = y;
      n2++;
    }
    if (n2 == 0) { n2 = 1; sum2 = 1.f; lastY = M; }
    float L = M + __logf(sum2);
    tau[b] = lastY;
    lse[b] = L;
    s_f[1] = L;
    s_i[0] = n2;
  }
  __syncthreads();
  const float L = s_f[1];
  const int n2 = s_i[0];

  // gumbel argmax over kept prefix
  float best = -__builtin_inff();
  int bestIdx = 0;
  for (int i = t; i < n2; i += 256) {
    unsigned long long c = cand[i];
    unsigned key = (unsigned)(c >> 32);
    unsigned idx = (unsigned)c;
    unsigned u = (key & 0x80000000u) ? (key & 0x7fffffffu) : ~key;
    float y = __uint_as_float(u);
    unsigned h = ((unsigned)b * 2654435761u) ^ (idx * 2246822519u);
    h ^= h >> 16; h *= 2654435761u; h ^= h >> 13;
    float uu = ((float)h + 0.5f) * (1.0f / 4294967296.0f);
    float g = -__logf(-__logf(uu));
    float sc = (y - L) + g;
    if (sc > best) { best = sc; bestIdx = (int)idx; }
  }
  sred[t] = best;
  sidx[t] = bestIdx;
  __syncthreads();
  for (int s = 128; s > 0; s >>= 1) {
    if (t < s && sred[t + s] > sred[t]) {
      sred[t] = sred[t + s];
      sidx[t] = sidx[t + s];
    }
    __syncthreads();
  }
  if (t == 0) tokens[b] = sidx[0];
}

// ---------------------------------------------------------------------------
// Kernel 4: logprobs writeout: y >= tau ? y - lse : -inf  (float4 streaming)
// grid = (V/1024, B), block 256, one float4 per thread.
// ---------------------------------------------------------------------------
__global__ __launch_bounds__(256) void finalize_kernel(
    const float* __restrict__ logits, const float* __restrict__ temps,
    const float* __restrict__ tau, const float* __restrict__ lse,
    float* __restrict__ out) {
  const int b = blockIdx.y;
  const int v4 = blockIdx.x * 256 + threadIdx.x;  // float4 index within row
  const float invT = 1.f / temps[b];
  const float tv = tau[b], l = lse[b];
  const float NEGINF = -__builtin_inff();
  float4 in = ((const float4*)(logits + (size_t)b * V_))[v4];
  float4 o;
  float yx = in.x * invT; o.x = (yx >= tv) ? yx - l : NEGINF;
  float yy = in.y * invT; o.y = (yy >= tv) ? yy - l : NEGINF;
  float yz = in.z * invT; o.z = (yz >= tv) ? yz - l : NEGINF;
  float yw = in.w * invT; o.w = (yw >= tv) ? yw - l : NEGINF;
  ((float4*)(out + (size_t)b * V_))[v4] = o;
}

// ---------------------------------------------------------------------------
extern "C" void kernel_launch(void* const* d_in, const int* in_sizes, int n_in,
                              void* d_out, int out_size, void* d_ws,
                              size_t ws_size, hipStream_t stream) {
  const float* hidden = (const float*)d_in[0];
  const float* emb    = (const float*)d_in[1];
  const float* bias   = (const float*)d_in[2];
  const int*   ids    = (const int*)d_in[3];
  const float* pres   = (const float*)d_in[4];
  const float* freq   = (const float*)d_in[5];
  const float* rep    = (const float*)d_in[6];
  const float* temps  = (const float*)d_in[7];
  const float* topps  = (const float*)d_in[8];
  const int*   topks  = (const int*)d_in[9];
  const float* minps  = (const float*)d_in[10];

  float* logits = (float*)d_ws;                       // B*V f32
  float* tau    = logits + (size_t)B_ * V_;           // B f32
  float* lsev   = tau + B_;                           // B f32

  float* out_lp  = (float*)d_out;                     // B*V logprobs
  int*   out_tok = (int*)((float*)d_out + (size_t)B_ * V_);  // B int32

  dim3 ggrid(V_ / 256, B_ / 16);
  gemm_kernel<<<ggrid, 256, 0, stream>>>(hidden, emb, bias, logits);

  penalty_kernel<<<B_, 256, 0, stream>>>(logits, ids, pres, freq, rep);

  rowselect_kernel<<<B_, 256, CAND * sizeof(unsigned long long), stream>>>(
      logits, temps, topps, topks, minps, tau, lsev, out_tok);

  dim3 fgrid(V_ / 1024, B_);
  finalize_kernel<<<fgrid, 256, 0, stream>>>(logits, temps, tau, lsev, out_lp);
}
